// BertSelfAttention_88124138979795
// MI455X (gfx1250) — compile-verified, matrix operand order checked by reading, not run on hardware
//
#include <hip/hip_runtime.h>

// ---------------------------------------------------------------------------
// BERT self-attention (relative_key_query) fused for gfx1250 (CDNA5, wave32).
// B=4 L=1024 DM=1024 H=16 D=64. All matrix math on v_wmma_f32_16x16x32_bf16.
// Flash-style: scores never touch HBM; Toeplitz position bias computed as
// band-GEMMs (Q*E_band^T, K*E_band^T) + LDS gathers. Software-pipelined
// fragment loads so WMMAs overlap outstanding global loads.
// ---------------------------------------------------------------------------

typedef __attribute__((ext_vector_type(16))) __bf16          v16bf;
typedef __attribute__((ext_vector_type(8)))  float           v8f;
typedef __attribute__((ext_vector_type(8)))  unsigned short  v8us;

#define BB   4
#define LL   1024
#define DM   1024
#define HH   16
#define DD   64
#define NPE  2047          // 2*MAXPOS-1 rows of dist_emb

static __device__ __forceinline__ unsigned short f2bf(float f) {
  unsigned u = __float_as_uint(f);
  u += 0x7FFFu + ((u >> 16) & 1u);            // round-to-nearest-even
  return (unsigned short)(u >> 16);
}
static __device__ __forceinline__ float bf2f(unsigned short s) {
  return __uint_as_float(((unsigned)s) << 16);
}

static __device__ __forceinline__ v8f wmma_bf16(v16bf a, v16bf b, v8f c) {
  // D(16x16,f32) = A(16x32,bf16) * B(32x16,bf16) + C
  return __builtin_amdgcn_wmma_f32_16x16x32_bf16(
      /*neg_a=*/false, a, /*neg_b=*/false, b,
      /*c_mod=*/(short)0, c, /*reuse_a=*/false, /*reuse_b=*/false);
}

// Load a 16x32 bf16 operand fragment from a row-major matrix (rows = the
// 16-wide tile dim, cols = K). ISA layout: lanes 0-15 hold M=lane, K groups
// {0..7,16..23}; lanes 16-31 hold same rows, K {8..15,24..31}.
// Used for A operands directly, and for B operands by passing B^T row-major.
static __device__ __forceinline__ v16bf load_frag(const unsigned short* base,
                                                  int row, int ld, int k0) {
  int lane = threadIdx.x & 31;
  int hlf  = lane >> 4;
  int m    = lane & 15;
  const unsigned short* p = base + (size_t)(row + m) * ld + k0 + hlf * 8;
  union { v16bf v; v8us h[2]; } u;
  u.h[0] = *(const v8us*)(p);
  u.h[1] = *(const v8us*)(p + 16);
  return u.v;
}

static __device__ __forceinline__ v16bf load_frag_clamp(const unsigned short* base,
                                                        int row, int ld, int k0,
                                                        int rmax) {
  int lane = threadIdx.x & 31;
  int hlf  = lane >> 4;
  int m    = lane & 15;
  int r    = row + m; if (r > rmax) r = rmax;
  const unsigned short* p = base + (size_t)r * ld + k0 + hlf * 8;
  union { v16bf v; v8us h[2]; } u;
  u.h[0] = *(const v8us*)(p);
  u.h[1] = *(const v8us*)(p + 16);
  return u.v;
}

// ------------------------------- conversion --------------------------------
__global__ void cvt_f32_bf16_k(const float* __restrict__ src,
                               unsigned short* __restrict__ dst, int n) {
  int i = blockIdx.x * blockDim.x + threadIdx.x;
  if (i < n) dst[i] = f2bf(src[i]);
}

// ------------------------------- QKV GEMM ----------------------------------
// out = X @ W^T + b. X: [4096,1024] bf16. W (row-major [j,k]) IS B^T already.
// blockIdx.z selects Q/K/V. Q,K stored [B,H,L,D]; V stored transposed [B,H,D,L]
// so that attention's P*V B-fragments load contiguously.
// k-loop is double-buffered: stage (n+1) loads issue before stage-n WMMAs.
__global__ __launch_bounds__(128) void qkv_gemm_k(
    const unsigned short* __restrict__ Xbf,
    const unsigned short* __restrict__ Wbf,   // 3 matrices back-to-back
    const float* __restrict__ bq, const float* __restrict__ bk,
    const float* __restrict__ bv,
    unsigned short* __restrict__ Qo, unsigned short* __restrict__ Ko,
    unsigned short* __restrict__ Vto) {
  const int w    = threadIdx.x >> 5;
  const int lane = threadIdx.x & 31;
  const int hlf  = lane >> 4, nn = lane & 15;
  const int z    = blockIdx.z;
  const int m0   = blockIdx.x * 64 + w * 16;   // output rows (b*L+l)
  const int n0   = blockIdx.y * 64;            // output cols (h*D+d)
  const unsigned short* W = Wbf + (size_t)z * DM * DM;
  const float* bias = (z == 0) ? bq : ((z == 1) ? bk : bv);

  v8f c[4] = {};

  // stage 0 preload
  v16bf a0 = load_frag(Xbf, m0, DM, 0);
  v16bf b0[4], b1[4];
#pragma unroll
  for (int j = 0; j < 4; ++j) b0[j] = load_frag(W, n0 + 16 * j, DM, 0);

  for (int kk = 0; kk < DM; kk += 64) {
    __builtin_prefetch(Xbf + (size_t)m0 * DM + kk + 512, 0, 1); // global_prefetch_b8
    // issue stage-1 loads, then consume stage-0 registers
    v16bf a1 = load_frag(Xbf, m0, DM, kk + 32);
#pragma unroll
    for (int j = 0; j < 4; ++j) b1[j] = load_frag(W, n0 + 16 * j, DM, kk + 32);
#pragma unroll
    for (int j = 0; j < 4; ++j) c[j] = wmma_bf16(a0, b0[j], c[j]);

    if (kk + 64 < DM) {
      a0 = load_frag(Xbf, m0, DM, kk + 64);
#pragma unroll
      for (int j = 0; j < 4; ++j) b0[j] = load_frag(W, n0 + 16 * j, DM, kk + 64);
    }
#pragma unroll
    for (int j = 0; j < 4; ++j) c[j] = wmma_bf16(a1, b1[j], c[j]);
  }

#pragma unroll
  for (int j = 0; j < 4; ++j) {
    const int   jg = n0 + 16 * j + nn;
    const float bz = bias[jg];
    const int   h  = jg >> 6, d = jg & 63;
#pragma unroll
    for (int i = 0; i < 8; ++i) {
      const int ig = m0 + i + 8 * hlf;        // global row
      const int bi = ig >> 10, l = ig & 1023;
      const int bh = bi * HH + h;
      const unsigned short val = f2bf(c[j][i] + bz);
      if (z == 0)      Qo [((size_t)bh * LL + l) * DD + d] = val;
      else if (z == 1) Ko [((size_t)bh * LL + l) * DD + d] = val;
      else             Vto[((size_t)bh * DD + d) * LL + l] = val;
    }
  }
}

// --------------------------- fused attention -------------------------------
// grid = (L/64 row tiles, B*H). 128 threads = 4 waves; wave w owns rows
// [l0+16w, l0+16w+16). Online softmax over 64-column chunks.
__global__ __launch_bounds__(128) void attn_fused_k(
    const unsigned short* __restrict__ Q,    // [B,H,L,D] bf16
    const unsigned short* __restrict__ K,    // [B,H,L,D] bf16
    const unsigned short* __restrict__ Vt,   // [B,H,D,L] bf16
    const unsigned short* __restrict__ PE,   // [2047,64] bf16
    const float* __restrict__ amask,         // [B,1,1,L]
    float* __restrict__ out) {               // [B,L,DM]
  __shared__ unsigned short QEs[4][16][128]; // per-wave Q*E band rows (bf16)
  __shared__ unsigned short KEs[64][128];    // shared K*E band rows (bf16)
  __shared__ unsigned short Ps [4][16][64];  // per-wave P restage (bf16)

  const int w    = threadIdx.x >> 5;
  const int lane = threadIdx.x & 31;
  const int hlf  = lane >> 4, nn = lane & 15;
  const int l0   = blockIdx.x * 64;
  const int bh   = blockIdx.y;
  const int b    = bh >> 4, h = bh & 15;

  const unsigned short* Qh = Q  + (size_t)bh * LL * DD;
  const unsigned short* Kh = K  + (size_t)bh * LL * DD;
  const unsigned short* Vh = Vt + (size_t)bh * DD * LL;

  const v16bf qa0 = load_frag(Qh, l0 + 16 * w, DD, 0);
  const v16bf qa1 = load_frag(Qh, l0 + 16 * w, DD, 32);

  v8f   o[4] = {};
  float mrow[8], srow[8];
#pragma unroll
  for (int i = 0; i < 8; ++i) { mrow[i] = -1e30f; srow[i] = 0.f; }

  for (int r0 = 0; r0 < LL; r0 += 64) {
    const int e_base = l0 - r0 + 960;        // band start in dist_emb

    // ---- S = Q K^T: issue ALL 8 K-fragment loads, then 8 WMMAs ---------
    v16bf kb[4][2];
#pragma unroll
    for (int j = 0; j < 4; ++j) {
      kb[j][0] = load_frag(Kh, r0 + 16 * j, DD, 0);
      kb[j][1] = load_frag(Kh, r0 + 16 * j, DD, 32);
    }
    v8f s[4];
#pragma unroll
    for (int j = 0; j < 4; ++j) {
      v8f c = {};
      c = wmma_bf16(qa0, kb[j][0], c);
      c = wmma_bf16(qa1, kb[j][1], c);
      s[j] = c;
    }

    // ---- band GEMMs: QE = Q*E^T (private), KE = K*E^T (shared) ---------
    // Double-buffered over band tiles t: preload t+1 while computing t.
    v16bf e0 = load_frag_clamp(PE, e_base, DD, 0,  NPE - 1);
    v16bf e1 = load_frag_clamp(PE, e_base, DD, 32, NPE - 1);
#pragma unroll
    for (int t = 0; t < 8; ++t) {
      v16bf f0 = e0, f1 = e1;
      if (t < 7) {
        e0 = load_frag_clamp(PE, e_base + 16 * (t + 1), DD, 0,  NPE - 1);
        e1 = load_frag_clamp(PE, e_base + 16 * (t + 1), DD, 32, NPE - 1);
      }
      v8f cq = {}, ck = {};
      cq = wmma_bf16(qa0,      f0, cq);
      cq = wmma_bf16(qa1,      f1, cq);
      ck = wmma_bf16(kb[w][0], f0, ck);
      ck = wmma_bf16(kb[w][1], f1, ck);
#pragma unroll
      for (int i = 0; i < 8; ++i) {
        const int m = i + 8 * hlf;
        QEs[w][m][16 * t + nn]       = f2bf(cq[i]);
        KEs[16 * w + m][16 * t + nn] = f2bf(ck[i]);
      }
    }
    __syncthreads();

    // ---- fuse bias + scale + mask into S -------------------------------
#pragma unroll
    for (int j = 0; j < 4; ++j) {
      const float mk = amask[b * LL + r0 + 16 * j + nn];
#pragma unroll
      for (int i = 0; i < 8; ++i) {
        const int m = i + 8 * hlf;
        const int t = 16 * w + m - 16 * j - nn + 63;   // in [0,127)
        const float qe = bf2f(QEs[w][m][t]);
        const float ke = bf2f(KEs[16 * j + nn][t]);
        s[j][i] = (s[j][i] + qe + ke) * 0.125f + mk;
      }
    }

    // ---- online softmax (rows live in 16-lane halves) ------------------
#pragma unroll
    for (int i = 0; i < 8; ++i) {
      float mnew = mrow[i];
#pragma unroll
      for (int j = 0; j < 4; ++j) mnew = fmaxf(mnew, s[j][i]);
#pragma unroll
      for (int off = 1; off < 16; off <<= 1)
        mnew = fmaxf(mnew, __shfl_xor(mnew, off, 32));
      const float alpha = __expf(mrow[i] - mnew);
      mrow[i] = mnew;
      float rs = 0.f;
#pragma unroll
      for (int j = 0; j < 4; ++j) {
        const float p = __expf(s[j][i] - mnew);
        s[j][i] = p;
        rs += p;
      }
#pragma unroll
      for (int off = 1; off < 16; off <<= 1)
        rs += __shfl_xor(rs, off, 32);
      srow[i] = srow[i] * alpha + rs;
#pragma unroll
      for (int jd = 0; jd < 4; ++jd) o[jd][i] *= alpha;
      const int m = i + 8 * hlf;
#pragma unroll
      for (int j = 0; j < 4; ++j) Ps[w][m][16 * j + nn] = f2bf(s[j][i]);
    }

    // ---- O += P * V: issue ALL V loads + P restage, then 8 WMMAs -------
    v16bf vb[4][2];
#pragma unroll
    for (int jd = 0; jd < 4; ++jd) {
      vb[jd][0] = load_frag(Vh, 16 * jd, LL, r0);      // B^T = Vt rows (d-major)
      vb[jd][1] = load_frag(Vh, 16 * jd, LL, r0 + 32);
    }
    const v16bf pa0 = load_frag(&Ps[w][0][0], 0, 64, 0);
    const v16bf pa1 = load_frag(&Ps[w][0][0], 0, 64, 32);
#pragma unroll
    for (int jd = 0; jd < 4; ++jd) {
      o[jd] = wmma_bf16(pa0, vb[jd][0], o[jd]);
      o[jd] = wmma_bf16(pa1, vb[jd][1], o[jd]);
    }
    __syncthreads();   // protect KEs before next chunk overwrites it
  }

  // ---- epilogue: normalize and scatter to [B,L,DM] ---------------------
#pragma unroll
  for (int i = 0; i < 8; ++i) {
    const float inv = 1.f / srow[i];
    const int   m   = i + 8 * hlf;
    const int   l   = l0 + 16 * w + m;
#pragma unroll
    for (int jd = 0; jd < 4; ++jd) {
      out[((size_t)(b * LL + l)) * DM + h * DD + 16 * jd + nn] = o[jd][i] * inv;
    }
  }
}

// ------------------------------- launcher ----------------------------------
extern "C" void kernel_launch(void* const* d_in, const int* in_sizes, int n_in,
                              void* d_out, int out_size, void* d_ws, size_t ws_size,
                              hipStream_t stream) {
  const float* X    = (const float*)d_in[0];
  const float* mask = (const float*)d_in[1];
  const float* Wq   = (const float*)d_in[2];
  const float* bq   = (const float*)d_in[3];
  const float* Wk   = (const float*)d_in[4];
  const float* bk   = (const float*)d_in[5];
  const float* Wv   = (const float*)d_in[6];
  const float* bv   = (const float*)d_in[7];
  const float* pe   = (const float*)d_in[8];
  float* out        = (float*)d_out;

  unsigned short* ws   = (unsigned short*)d_ws;
  unsigned short* Xbf  = ws;                             // 4096*1024
  unsigned short* Wbf  = Xbf  + (size_t)4096 * 1024;     // 3*1024*1024
  unsigned short* PEbf = Wbf  + (size_t)3 * 1024 * 1024; // 2047*64 (padded)
  unsigned short* Qbf  = PEbf + 131072;                  // B*H*L*D
  unsigned short* Kbf  = Qbf  + (size_t)BB * HH * LL * DD;
  unsigned short* Vtbf = Kbf  + (size_t)BB * HH * LL * DD;

  const int NX = 4096 * 1024, NW = 1024 * 1024, NE = NPE * DD;
  cvt_f32_bf16_k<<<(NX + 255) / 256, 256, 0, stream>>>(X,  Xbf,            NX);
  cvt_f32_bf16_k<<<(NW + 255) / 256, 256, 0, stream>>>(Wq, Wbf,            NW);
  cvt_f32_bf16_k<<<(NW + 255) / 256, 256, 0, stream>>>(Wk, Wbf + NW,       NW);
  cvt_f32_bf16_k<<<(NW + 255) / 256, 256, 0, stream>>>(Wv, Wbf + 2 * NW,   NW);
  cvt_f32_bf16_k<<<(NE + 255) / 256, 256, 0, stream>>>(pe, PEbf,           NE);

  dim3 g1(64, 16, 3);   // 4096/64 row tiles, 1024/64 col tiles, {Q,K,V}
  qkv_gemm_k<<<g1, 128, 0, stream>>>(Xbf, Wbf, bq, bk, bv, Qbf, Kbf, Vtbf);

  dim3 g2(16, 64);      // 1024/64 row tiles, B*H
  attn_fused_k<<<g2, 128, 0, stream>>>(Qbf, Kbf, Vtbf, PEbf, mask, out);
}
